// GraphDetector_77979426226958
// MI455X (gfx1250) — matrix operand order repaired and checked by reading, not compiled, not run on hardware
//
#include <hip/hip_runtime.h>
#include <hip/hip_bf16.h>

typedef __attribute__((ext_vector_type(16))) _Float16 v16h;
typedef __attribute__((ext_vector_type(8)))  float    v8f;

#define TT  8
#define NNODE 8192
#define DEGR 15
#define HHEAD 4
#define EDGE_T (NNODE * (DEGR + 1))   // 131072 edges per timestep

// ---------------------------------------------------------------------------
// Weight prep: transpose f32 [K, Nc] -> f16 [Nc, K] so each WMMA B-lane loads
// 16 contiguous f16 (32B) per K-step.
// ---------------------------------------------------------------------------
__global__ __launch_bounds__(256) void wprep_kernel(const float* __restrict__ w,
                                                    _Float16* __restrict__ wt,
                                                    int K, int Nc) {
    int idx = blockIdx.x * blockDim.x + threadIdx.x;
    if (idx >= K * Nc) return;
    int n = idx / K;
    int k = idx - n * K;
    wt[idx] = (_Float16)w[(size_t)k * Nc + n];
}

// ---------------------------------------------------------------------------
// WMMA GEMM: C[M,Nc] = act( A[M,K](f32->f16) * Bt[Nc,K](f16) + bias )
// 4 waves per block, each wave owns one 16x16 output tile.
// grid = (Nc/64, M/16), block = 128.
// act: 0 = none, 1 = ELU, 2 = tanh
// ---------------------------------------------------------------------------
__global__ __launch_bounds__(128) void wmma_gemm_kernel(
    const float* __restrict__ A, const _Float16* __restrict__ Bt,
    float* __restrict__ C, int M, int K, int Nc,
    const float* __restrict__ bias, int act) {
    const int lane = threadIdx.x & 31;
    const int wave = threadIdx.x >> 5;
    const int nt   = blockIdx.x * 4 + wave;   // 16-col tile index
    const int mt   = blockIdx.y;              // 16-row tile index
    const int half = lane >> 4;
    const int l15  = lane & 15;

    const float*    arow = A  + (size_t)(mt * 16 + l15) * K;
    const _Float16* bcol = Bt + (size_t)(nt * 16 + l15) * K + half * 16;

    v8f acc = {};
    for (int kb = 0; kb < K; kb += 32) {
        v16h av, bv;
        const float* a0 = arow + kb + half * 16;      // K = kb + half*16 .. +8
        const float* a1 = a0 + 8;                     // K = kb + half*16 + 8 .. +8
#pragma unroll
        for (int j = 0; j < 8; ++j) {
            av[j]     = (_Float16)a0[j];
            av[8 + j] = (_Float16)a1[j];
        }
        const _Float16* bp = bcol + kb;               // K = kb + half*16 .. +16
#pragma unroll
        for (int j = 0; j < 16; ++j) bv[j] = bp[j];
        acc = __builtin_amdgcn_wmma_f32_16x16x32_f16(
            false, av, false, bv, (short)0, acc, false, false);
    }

    const int ocol = nt * 16 + l15;
    const float badd = bias ? bias[ocol] : 0.0f;
#pragma unroll
    for (int r = 0; r < 8; ++r) {
        int orow = mt * 16 + half * 8 + r;
        float v = acc[r] + badd;
        if (act == 1)      v = v > 0.0f ? v : (__expf(v) - 1.0f);
        else if (act == 2) v = tanhf(v);
        C[(size_t)orow * Nc + ocol] = v;
    }
}

// ---------------------------------------------------------------------------
// GATv2 attention + softmax + head-mean + bias + ELU, fixed 16 in-edges/node.
// Node i's sources: src[i*15 + k] (k<15) and i itself (self loop).
// One block per node, 4 waves = 4 heads. Gathered gl rows cached in LDS and
// reused for the weighted sum; head mean reduced through LDS.
// ---------------------------------------------------------------------------
template <int C>
__global__ __launch_bounds__(128) void gat_attn_kernel(
    const float* __restrict__ gl, const float* __restrict__ gr,
    const float* __restrict__ att, const float* __restrict__ bias,
    const int* __restrict__ src,
    float* __restrict__ out, int outStride, int outCol0) {
    constexpr int CPL = C / 32;         // channels per lane
    constexpr int HC  = HHEAD * C;

    __shared__ float tileAll[HHEAD * 16 * C];   // gathered gl[src] rows per head
    __shared__ float hres[HHEAD * C];           // per-head results

    const int i    = blockIdx.x;
    const int h    = threadIdx.x >> 5;
    const int lane = threadIdx.x & 31;
    float* tile = tileAll + h * 16 * C;

    float grv[CPL], av[CPL];
    const float* grp = gr + (size_t)i * HC + h * C;
    const float* ap  = att + h * C;
#pragma unroll
    for (int j = 0; j < CPL; ++j) {
        int c = lane + 32 * j;
        grv[j] = grp[c];
        av[j]  = ap[c];
    }

    // pass 1: attention logits e_k = a . leaky_relu(gl[s] + gr[i])
    float e[16];
    const int* sp = src + (size_t)i * DEGR;
#pragma unroll
    for (int k = 0; k < 16; ++k) {
        int s = (k < DEGR) ? sp[k] : i;
        const float* glp = gl + (size_t)s * HC + h * C;
        float p = 0.0f;
#pragma unroll
        for (int j = 0; j < CPL; ++j) {
            int c = lane + 32 * j;
            float g = glp[c];
            tile[k * C + c] = g;
            float z = g + grv[j];
            z = z > 0.0f ? z : 0.2f * z;   // LeakyReLU(0.2)
            p += av[j] * z;
        }
#pragma unroll
        for (int off = 16; off > 0; off >>= 1) p += __shfl_xor(p, off, 32);
        e[k] = p;
    }

    // softmax over 16 incoming edges
    float m = e[0];
#pragma unroll
    for (int k = 1; k < 16; ++k) m = fmaxf(m, e[k]);
    float w[16], den = 0.0f;
#pragma unroll
    for (int k = 0; k < 16; ++k) { w[k] = __expf(e[k] - m); den += w[k]; }
    const float inv = 1.0f / den;

    // pass 2: weighted message sum from LDS tile
    float accv[CPL];
#pragma unroll
    for (int j = 0; j < CPL; ++j) accv[j] = 0.0f;
#pragma unroll
    for (int k = 0; k < 16; ++k)
#pragma unroll
        for (int j = 0; j < CPL; ++j)
            accv[j] += w[k] * tile[k * C + lane + 32 * j];
#pragma unroll
    for (int j = 0; j < CPL; ++j)
        hres[h * C + lane + 32 * j] = accv[j] * inv;
    __syncthreads();

    // head mean + bias + ELU (every GAT layer in this model is ELU'd)
    const int tid = threadIdx.x;
    if (tid < C) {
        float s = 0.0f;
#pragma unroll
        for (int hh = 0; hh < HHEAD; ++hh) s += hres[hh * C + tid];
        s = 0.25f * s + bias[tid];
        s = s > 0.0f ? s : (__expf(s) - 1.0f);
        out[(size_t)i * outStride + outCol0 + tid] = s;
    }
}

// dst[row*dstStride + dstOff + c] = elu(src[row*C + c])  (concat second half)
__global__ __launch_bounds__(256) void elu_copy_kernel(
    const float* __restrict__ srcv, float* __restrict__ dst,
    long total, int C, int dstStride, int dstOff) {
    long idx = (long)blockIdx.x * blockDim.x + threadIdx.x;
    if (idx >= total) return;
    long r = idx / C;
    int  c = (int)(idx - r * C);
    float v = srcv[idx];
    v = v > 0.0f ? v : (__expf(v) - 1.0f);
    dst[r * (long)dstStride + dstOff + c] = v;
}

// ---------------------------------------------------------------------------
extern "C" void kernel_launch(void* const* d_in, const int* in_sizes, int n_in,
                              void* d_out, int out_size, void* d_ws, size_t ws_size,
                              hipStream_t stream) {
    (void)in_sizes; (void)n_in; (void)out_size; (void)ws_size;
    const float* x    = (const float*)d_in[0];
    const int*   esrc = (const int*)d_in[1];

    // ---- workspace carve-out ----------------------------------------------
    char*  wsb = (char*)d_ws;
    size_t off = 0;
    auto alloc = [&](size_t bytes) -> void* {
        void* p = wsb + off;
        off = (off + bytes + 255) & ~(size_t)255;
        return p;
    };

    // f16 transposed weights: {d_in idx, K(cin), Nc(H*cout or cout)}
    struct WSpec { int idx; int K; int Nc; };
    const WSpec sp[22] = {
        {3,64,512},{4,64,512},      // c1 wl,wr
        {7,128,256},{8,128,256},    // c2
        {11,64,256},{12,64,256},    // c3
        {15,128,256},{16,128,256},  // c4
        {19,64,256},{20,64,256},    // r11
        {23,64,256},{24,64,256},    // r12
        {27,128,512},{28,128,512},  // r2
        {31,64,256},{32,64,256},    // f11
        {35,128,512},{36,128,512},  // f2
        {39,128,64},                // r3
        {41,128,64},                // f3
        {43,64,64},                 // r4
        {45,64,64},                 // f4
    };
    _Float16* wt[22];
    for (int s = 0; s < 22; ++s) {
        int cnt = sp[s].K * sp[s].Nc;
        wt[s] = (_Float16*)alloc((size_t)cnt * sizeof(_Float16));
        wprep_kernel<<<(cnt + 255) / 256, 256, 0, stream>>>(
            (const float*)d_in[sp[s].idx], wt[s], sp[s].K, sp[s].Nc);
    }

    float* GL = (float*)alloc((size_t)NNODE * 512 * sizeof(float));
    float* GR = (float*)alloc((size_t)NNODE * 512 * sizeof(float));
    float* H1 = (float*)alloc((size_t)TT * NNODE * 128 * sizeof(float)); // reused as H3
    float* H2 = (float*)alloc((size_t)TT * NNODE * 64 * sizeof(float));
    float* BA = (float*)alloc((size_t)NNODE * 64  * sizeof(float));
    float* BB = (float*)alloc((size_t)NNODE * 128 * sizeof(float));
    float* BC = (float*)alloc((size_t)NNODE * 128 * sizeof(float));
    float* BD = (float*)alloc((size_t)NNODE * 64  * sizeof(float));

    float* RECON = (float*)d_out;             // [N,64]
    float* FORE  = RECON + (size_t)NNODE * 64; // [N,64]
    float* EMB   = FORE  + (size_t)NNODE * 64; // [T,N,64]

    // ---- one GATv2 layer: gl/gr GEMMs + attention --------------------------
    auto gat_layer = [&](const float* xin, int Cin, int Cout,
                         const _Float16* wlT, const _Float16* wrT,
                         const float* aP, const float* bP,
                         const int* srcP, float* outP, int outStride, int outCol0) {
        const int HC = HHEAD * Cout;
        dim3 gg(HC / 64, NNODE / 16);
        wmma_gemm_kernel<<<gg, 128, 0, stream>>>(xin, wlT, GL, NNODE, Cin, HC, nullptr, 0);
        wmma_gemm_kernel<<<gg, 128, 0, stream>>>(xin, wrT, GR, NNODE, Cin, HC, nullptr, 0);
        if (Cout == 64)
            gat_attn_kernel<64><<<NNODE, 128, 0, stream>>>(GL, GR, aP, bP, srcP,
                                                           outP, outStride, outCol0);
        else
            gat_attn_kernel<128><<<NNODE, 128, 0, stream>>>(GL, GR, aP, bP, srcP,
                                                            outP, outStride, outCol0);
    };
    auto F = [&](int i) { return (const float*)d_in[i]; };

    // ---- spatio-temporal encoder (per-timestep GATv2, shared weights) ------
    for (int t = 0; t < TT; ++t)    // c1: [T,N,64] -> H1 [T,N,128]
        gat_layer(x + (size_t)t * NNODE * 64, 64, 128, wt[0], wt[1], F(5), F(6),
                  esrc + (size_t)t * EDGE_T, H1 + (size_t)t * NNODE * 128, 128, 0);
    for (int t = 0; t < TT; ++t)    // c2: H1 -> H2 [T,N,64]
        gat_layer(H1 + (size_t)t * NNODE * 128, 128, 64, wt[2], wt[3], F(9), F(10),
                  esrc + (size_t)t * EDGE_T, H2 + (size_t)t * NNODE * 64, 64, 0);
    for (int t = 0; t < TT; ++t)    // c3: H2 -> H1[:, :64] (H1 reused as concat buf)
        gat_layer(H2 + (size_t)t * NNODE * 64, 64, 64, wt[4], wt[5], F(13), F(14),
                  esrc + (size_t)t * EDGE_T, H1 + (size_t)t * NNODE * 128, 128, 0);
    {   // H1[:, 64:128] = elu(H2)
        long total = (long)TT * NNODE * 64;
        elu_copy_kernel<<<(unsigned)((total + 255) / 256), 256, 0, stream>>>(
            H2, H1, total, 64, 128, 64);
    }
    for (int t = 0; t < TT; ++t)    // c4: H1 -> EMB (d_out) [T,N,64]
        gat_layer(H1 + (size_t)t * NNODE * 128, 128, 64, wt[6], wt[7], F(17), F(18),
                  esrc + (size_t)t * EDGE_T, EMB + (size_t)t * NNODE * 64, 64, 0);

    // ---- reconstruction branch (t = T-1) -----------------------------------
    const int*   s7 = esrc + (size_t)(TT - 1) * EDGE_T;
    const float* e7 = EMB + (size_t)(TT - 1) * NNODE * 64;
    gat_layer(e7, 64, 64, wt[8],  wt[9],  F(21), F(22), s7, BA, 64, 0);      // r11
    gat_layer(BA, 64, 64, wt[10], wt[11], F(25), F(26), s7, BB, 128, 0);     // r12
    {
        long total = (long)NNODE * 64;
        elu_copy_kernel<<<(unsigned)((total + 255) / 256), 256, 0, stream>>>(
            BA, BB, total, 64, 128, 64);
    }
    gat_layer(BB, 128, 128, wt[12], wt[13], F(29), F(30), s7, BC, 128, 0);   // r2
    wmma_gemm_kernel<<<dim3(1, NNODE / 16), 128, 0, stream>>>(
        BC, wt[18], BD, NNODE, 128, 64, F(40), 2);                           // tanh(.@r3+b)
    wmma_gemm_kernel<<<dim3(1, NNODE / 16), 128, 0, stream>>>(
        BD, wt[20], RECON, NNODE, 64, 64, F(44), 0);                         // recon

    // ---- forecasting branch (t = T-2) --------------------------------------
    const int*   s6 = esrc + (size_t)(TT - 2) * EDGE_T;
    const float* e6 = EMB + (size_t)(TT - 2) * NNODE * 64;
    gat_layer(e6, 64, 64, wt[14], wt[15], F(33), F(34), s6, BA, 64, 0);      // f11
    gat_layer(BA, 64, 64, wt[10], wt[11], F(25), F(26), s6, BB, 128, 0);     // r12 (shared)
    {
        long total = (long)NNODE * 64;
        elu_copy_kernel<<<(unsigned)((total + 255) / 256), 256, 0, stream>>>(
            BA, BB, total, 64, 128, 64);
    }
    gat_layer(BB, 128, 128, wt[16], wt[17], F(37), F(38), s6, BC, 128, 0);   // f2
    wmma_gemm_kernel<<<dim3(1, NNODE / 16), 128, 0, stream>>>(
        BC, wt[19], BD, NNODE, 128, 64, F(42), 2);                           // tanh(.@f3+b)
    wmma_gemm_kernel<<<dim3(1, NNODE / 16), 128, 0, stream>>>(
        BD, wt[21], FORE, NNODE, 64, 64, F(46), 0);                          // forecast
}
